// SimplifiedCrossAttention_20151986553032
// MI455X (gfx1250) — matrix operand order repaired and checked by reading, not compiled, not run on hardware
//
#include <hip/hip_runtime.h>

// ---------------- problem constants (from reference) ----------------
#define BSZ   4
#define NVIS  4096
#define NTXT  1024
#define VDIM  1024
#define TDIM  1024
#define HEADS 8
#define DHEAD 128   // TDIM / HEADS

// ---------------- WMMA types ----------------
typedef __attribute__((ext_vector_type(16))) __bf16 v16bf;
typedef __attribute__((ext_vector_type(8)))  float  v8f;

union BF16Frag {
    v16bf          v;
    uint4          q[2];
    unsigned short u[16];
};

__device__ __forceinline__ unsigned short f2bf(float f) {
    unsigned int u = __float_as_uint(f);
    u += 0x7FFFu + ((u >> 16) & 1u);   // round-to-nearest-even
    return (unsigned short)(u >> 16);
}

__device__ __forceinline__ unsigned pk2bf(float lo, float hi) {
    return (unsigned)f2bf(lo) | ((unsigned)f2bf(hi) << 16);
}

// A fragment: lane row = lane&15; halves at K = (lane>>4)*8 and +16
#define LDA(dst, base, KB)                                          \
    do {                                                            \
        (dst).q[0] = *(const uint4*)((base) + (KB) + akoff);        \
        (dst).q[1] = *(const uint4*)((base) + (KB) + akoff + 16);   \
    } while (0)

// B fragment: lane col = lane&15; halves at K = (lane>>4)*8 and +16
#define LDB(dst, base, KB)                                          \
    do {                                                            \
        (dst).q[0] = *(const uint4*)((base) + (KB) + bkoff);        \
        (dst).q[1] = *(const uint4*)((base) + (KB) + bkoff + 16);   \
    } while (0)

#define WMMA(acc, a, b)                                             \
    (acc) = __builtin_amdgcn_wmma_f32_16x16x32_bf16(                \
        false, (a).v, false, (b).v, (short)0, (acc), false, false)

// ---------------- kernel 1: f32 -> packed bf16 convert ----------------
__global__ __launch_bounds__(256) void cvt_f32_bf16_v4(const float4* __restrict__ src,
                                                       uint2* __restrict__ dst, long n4) {
    long i = (long)blockIdx.x * blockDim.x + threadIdx.x;
    long stride = (long)gridDim.x * blockDim.x;
    for (; i < n4; i += stride) {
        float4 x = src[i];
        uint2 o;
        o.x = pk2bf(x.x, x.y);
        o.y = pk2bf(x.z, x.w);
        dst[i] = o;
    }
}

// ---------------- kernel 2: vc = visual @ W^T (bf16 WMMA, dual-layout store) ----
// A: visual_bf [16384 x 1024] row-major (K contiguous)
// B: W_bf [1024 x 1024] row-major; Bmat[k][t] = W[t][k] -> lane col t, contiguous K
// C: vc [16384 x 1024] bf16 (scatter b16) and vc_t [4][1024][4096] bf16 (packed b128)
__global__ __launch_bounds__(256) void gemm_vc(const unsigned short* __restrict__ Abf,
                                               const unsigned short* __restrict__ Wbf,
                                               unsigned short* __restrict__ vc,
                                               unsigned short* __restrict__ vct) {
    const int lane  = threadIdx.x & 31;
    const int w     = blockIdx.x * 8 + (threadIdx.x >> 5);   // 16384 waves
    const int mtile = w >> 4;                                // 0..1023
    const int ngrp  = w & 15;                                // 64 cols each
    const int ln    = lane & 15;
    const int hi    = lane >> 4;
    const int akoff = hi * 8;
    const int bkoff = hi * 8;

    const unsigned short* arow = Abf + (long)(mtile * 16 + ln) * VDIM;
    const unsigned short* bcol[4];
#pragma unroll
    for (int nt = 0; nt < 4; ++nt)
        bcol[nt] = Wbf + (long)(ngrp * 64 + nt * 16 + ln) * VDIM;

    v8f acc[4] = {};

    // software pipeline: double-buffered fragments
    BF16Frag a0, b0[4];
    LDA(a0, arow, 0);
#pragma unroll
    for (int nt = 0; nt < 4; ++nt) LDB(b0[nt], bcol[nt], 0);

#pragma unroll 2
    for (int kb = 32; kb < VDIM; kb += 32) {
        // L2 prefetch for the cold visual A-stream, ~1KB ahead (clamped in-row)
        const int kpf = (kb + 512 < VDIM) ? kb + 512 : kb;
        __builtin_prefetch(arow + kpf, 0, 1);

        BF16Frag a1, b1[4];
        LDA(a1, arow, kb);
#pragma unroll
        for (int nt = 0; nt < 4; ++nt) LDB(b1[nt], bcol[nt], kb);
#pragma unroll
        for (int nt = 0; nt < 4; ++nt) WMMA(acc[nt], a0, b0[nt]);
        a0 = a1;
#pragma unroll
        for (int nt = 0; nt < 4; ++nt) b0[nt] = b1[nt];
    }
#pragma unroll
    for (int nt = 0; nt < 4; ++nt) WMMA(acc[nt], a0, b0[nt]);

    // stores: vct packs 8 consecutive j per lane into one b128
    const int bb = mtile >> 8;                        // batch
    const int j0 = (mtile & 255) * 16 + hi * 8;       // first j of this lane's rows
#pragma unroll
    for (int nt = 0; nt < 4; ++nt) {
        const int t = ngrp * 64 + nt * 16 + ln;
        uint4 pk;
        pk.x = pk2bf(acc[nt][0], acc[nt][1]);
        pk.y = pk2bf(acc[nt][2], acc[nt][3]);
        pk.z = pk2bf(acc[nt][4], acc[nt][5]);
        pk.w = pk2bf(acc[nt][6], acc[nt][7]);
        *(uint4*)(vct + ((long)bb * TDIM + t) * NVIS + j0) = pk;
        const unsigned short* hp = (const unsigned short*)&pk;
#pragma unroll
        for (int r = 0; r < 8; ++r) {
            int m = mtile * 16 + hi * 8 + r;
            vc[(long)m * TDIM + t] = hp[r];
        }
    }
}

// ---------------- kernel 3: raw scores = q @ k^T per (b,h) ----------------
__global__ __launch_bounds__(256) void sim_gemm(const unsigned short* __restrict__ Tbf,
                                                const unsigned short* __restrict__ vc,
                                                float* __restrict__ scores) {
    const int lane   = threadIdx.x & 31;
    const int w      = blockIdx.x * 8 + (threadIdx.x >> 5);  // 131072 waves
    const int bh     = w >> 12;                              // 0..31
    const int within = w & 4095;
    const int mtile  = within >> 6;                          // 0..63 (i tiles)
    const int ngrp   = within & 63;                          // 64 j cols each
    const int b = bh >> 3, h = bh & 7;
    const int ln = lane & 15, hi = lane >> 4;
    const int akoff = hi * 8, bkoff = hi * 8;

    const unsigned short* arow = Tbf + ((long)b * NTXT + mtile * 16 + ln) * TDIM + h * DHEAD;
    const unsigned short* bcol[4];
#pragma unroll
    for (int nt = 0; nt < 4; ++nt) {
        int j = ngrp * 64 + nt * 16 + ln;
        bcol[nt] = vc + ((long)b * NVIS + j) * TDIM + h * DHEAD;
    }

    v8f acc[4] = {};
    BF16Frag a0, b0[4];
    LDA(a0, arow, 0);
#pragma unroll
    for (int nt = 0; nt < 4; ++nt) LDB(b0[nt], bcol[nt], 0);

#pragma unroll
    for (int kb = 32; kb < DHEAD; kb += 32) {
        BF16Frag a1, b1[4];
        LDA(a1, arow, kb);
#pragma unroll
        for (int nt = 0; nt < 4; ++nt) LDB(b1[nt], bcol[nt], kb);
#pragma unroll
        for (int nt = 0; nt < 4; ++nt) WMMA(acc[nt], a0, b0[nt]);
        a0 = a1;
#pragma unroll
        for (int nt = 0; nt < 4; ++nt) b0[nt] = b1[nt];
    }
#pragma unroll
    for (int nt = 0; nt < 4; ++nt) WMMA(acc[nt], a0, b0[nt]);

#pragma unroll
    for (int nt = 0; nt < 4; ++nt) {
#pragma unroll
        for (int r = 0; r < 8; ++r) {
            int i = mtile * 16 + hi * 8 + r;
            int j = ngrp * 64 + nt * 16 + ln;
            scores[((long)(b * HEADS + h) * NTXT + i) * NVIS + j] = acc[nt][r];
        }
    }
}

// ---------------- kernel 4: in-place row softmax, rows of 4096 ----------------
__global__ __launch_bounds__(256) void softmax_rows(float* __restrict__ attn) {
    float* p = attn + (long)blockIdx.x * NVIS;
    __shared__ float red[8];
    const int tid = threadIdx.x;

    float vals[16];
    float m = -INFINITY;
#pragma unroll
    for (int i = 0; i < 16; ++i) {
        vals[i] = p[tid + i * 256];
        m = fmaxf(m, vals[i]);
    }
#pragma unroll
    for (int off = 16; off > 0; off >>= 1) m = fmaxf(m, __shfl_xor(m, off, 32));
    if ((tid & 31) == 0) red[tid >> 5] = m;
    __syncthreads();
    float mx = red[0];
#pragma unroll
    for (int i = 1; i < 8; ++i) mx = fmaxf(mx, red[i]);
    __syncthreads();

    float s = 0.f;
#pragma unroll
    for (int i = 0; i < 16; ++i) {
        vals[i] = __expf(vals[i] - mx);
        s += vals[i];
    }
#pragma unroll
    for (int off = 16; off > 0; off >>= 1) s += __shfl_xor(s, off, 32);
    if ((tid & 31) == 0) red[tid >> 5] = s;
    __syncthreads();
    float sum = 0.f;
#pragma unroll
    for (int i = 0; i < 8; ++i) sum += red[i];
    float inv = 1.0f / sum;
#pragma unroll
    for (int i = 0; i < 16; ++i) p[tid + i * 256] = vals[i] * inv;
}

// ---------------- kernel 5: out = attn @ k per (b,h) ----------------
// A: attn f32 [1024 x 4096] (K = j contiguous), converted to bf16 at load
// B: vc_t [b][t][j] (K = j contiguous) -> lane col = t
// C: f32 out[b][i][h*128 + t]
__global__ __launch_bounds__(256) void out_gemm(const float* __restrict__ attn,
                                                const unsigned short* __restrict__ vct,
                                                float* __restrict__ out) {
    const int lane  = threadIdx.x & 31;
    const int w     = blockIdx.x * 8 + (threadIdx.x >> 5);   // 2048 waves
    const int bh    = w >> 6;                                // 0..31
    const int mtile = w & 63;                                // i tiles
    const int b = bh >> 3, h = bh & 7;
    const int ln = lane & 15, hi = lane >> 4;
    const int akoff = hi * 8, bkoff = hi * 8;

    const float* arow = attn + ((long)(b * HEADS + h) * NTXT + mtile * 16 + ln) * NVIS;
    const unsigned short* bcol[8];
#pragma unroll
    for (int nt = 0; nt < 8; ++nt) {
        int t = h * DHEAD + nt * 16 + ln;
        bcol[nt] = vct + ((long)b * TDIM + t) * NVIS;
    }

    v8f acc[8] = {};

    // prologue: step 0 A-floats and B-fragments
    float4 x0 = *(const float4*)(arow + akoff);
    float4 x1 = *(const float4*)(arow + akoff + 4);
    float4 x2 = *(const float4*)(arow + akoff + 16);
    float4 x3 = *(const float4*)(arow + akoff + 20);
    BF16Frag b0[8];
#pragma unroll
    for (int nt = 0; nt < 8; ++nt) LDB(b0[nt], bcol[nt], 0);

    BF16Frag a0;
    a0.q[0].x = pk2bf(x0.x, x0.y); a0.q[0].y = pk2bf(x0.z, x0.w);
    a0.q[0].z = pk2bf(x1.x, x1.y); a0.q[0].w = pk2bf(x1.z, x1.w);
    a0.q[1].x = pk2bf(x2.x, x2.y); a0.q[1].y = pk2bf(x2.z, x2.w);
    a0.q[1].z = pk2bf(x3.x, x3.y); a0.q[1].w = pk2bf(x3.z, x3.w);

#pragma unroll 2
    for (int kb = 32; kb < NVIS; kb += 32) {
        // attn is the only HBM-streaming operand (537MB > L2); prefetch ~1KB ahead
        const int kpf = (kb + 256 < NVIS) ? kb + 256 : kb;
        __builtin_prefetch(arow + kpf, 0, 1);

        const float* pa = arow + kb + akoff;
        float4 y0 = *(const float4*)(pa);
        float4 y1 = *(const float4*)(pa + 4);
        float4 y2 = *(const float4*)(pa + 16);
        float4 y3 = *(const float4*)(pa + 20);
        BF16Frag b1[8];
#pragma unroll
        for (int nt = 0; nt < 8; ++nt) LDB(b1[nt], bcol[nt], kb);

#pragma unroll
        for (int nt = 0; nt < 8; ++nt) WMMA(acc[nt], a0, b0[nt]);

        a0.q[0].x = pk2bf(y0.x, y0.y); a0.q[0].y = pk2bf(y0.z, y0.w);
        a0.q[0].z = pk2bf(y1.x, y1.y); a0.q[0].w = pk2bf(y1.z, y1.w);
        a0.q[1].x = pk2bf(y2.x, y2.y); a0.q[1].y = pk2bf(y2.z, y2.w);
        a0.q[1].z = pk2bf(y3.x, y3.y); a0.q[1].w = pk2bf(y3.z, y3.w);
#pragma unroll
        for (int nt = 0; nt < 8; ++nt) b0[nt] = b1[nt];
    }
#pragma unroll
    for (int nt = 0; nt < 8; ++nt) WMMA(acc[nt], a0, b0[nt]);

#pragma unroll
    for (int nt = 0; nt < 8; ++nt) {
#pragma unroll
        for (int r = 0; r < 8; ++r) {
            int i = mtile * 16 + hi * 8 + r;
            int t = h * DHEAD + nt * 16 + ln;
            out[((long)b * NTXT + i) * TDIM + t] = acc[nt][r];
        }
    }
}

// ---------------- host launcher ----------------
extern "C" void kernel_launch(void* const* d_in, const int* in_sizes, int n_in,
                              void* d_out, int out_size, void* d_ws, size_t ws_size,
                              hipStream_t stream) {
    (void)in_sizes; (void)n_in; (void)out_size; (void)ws_size;
    const float* visual  = (const float*)d_in[0];   // [4,4096,1024]
    const float* textual = (const float*)d_in[1];   // [4,1024,1024]
    const float* W       = (const float*)d_in[2];   // [1024,1024]

    float* out  = (float*)d_out;                        // [4,1024,1024]
    float* attn = out + (long)BSZ * NTXT * TDIM;        // [4,8,1024,4096]

    // workspace layout (bf16 = ushort), ~111 MB total
    unsigned short* visual_bf  = (unsigned short*)d_ws;                  // 16.7M elems
    unsigned short* textual_bf = visual_bf  + (long)BSZ * NVIS * VDIM;   // 4.2M
    unsigned short* W_bf       = textual_bf + (long)BSZ * NTXT * TDIM;   // 1.0M
    unsigned short* vc_bf      = W_bf       + (long)TDIM * VDIM;         // 16.7M
    unsigned short* vct_bf     = vc_bf      + (long)BSZ * NVIS * TDIM;   // 16.7M

    // 1) f32 -> bf16 converts
    cvt_f32_bf16_v4<<<4096, 256, 0, stream>>>((const float4*)visual,
                                              (uint2*)visual_bf,
                                              (long)BSZ * NVIS * VDIM / 4);
    cvt_f32_bf16_v4<<<2048, 256, 0, stream>>>((const float4*)textual,
                                              (uint2*)textual_bf,
                                              (long)BSZ * NTXT * TDIM / 4);
    cvt_f32_bf16_v4<<<512, 256, 0, stream>>>((const float4*)W,
                                             (uint2*)W_bf,
                                             (long)TDIM * VDIM / 4);

    // 2) projection GEMM: 16384 waves, 8 waves/block
    gemm_vc<<<2048, 256, 0, stream>>>(visual_bf, W_bf, vc_bf, vct_bf);

    // 3) raw attention scores: 131072 waves
    sim_gemm<<<16384, 256, 0, stream>>>(textual_bf, vc_bf, attn);

    // 4) softmax over 32768 rows of 4096
    softmax_rows<<<BSZ * HEADS * NTXT, 256, 0, stream>>>(attn);

    // 5) output GEMM: 2048 waves
    out_gemm<<<256, 256, 0, stream>>>(attn, vct_bf, out);
}
